// DualGATv2_3719441678809
// MI455X (gfx1250) — matrix operand order repaired and checked by reading, not compiled, not run on hardware
//
#include <hip/hip_runtime.h>
#include <hip/hip_fp16.h>

#define NEG_SLOPE 0.2f
#define LN_EPS 1e-5f

typedef __attribute__((ext_vector_type(16))) _Float16 v16h;
typedef __attribute__((ext_vector_type(8)))  _Float16 v8h;
typedef __attribute__((ext_vector_type(8)))  float    v8f;

// ---------------------------------------------------------------------------
// Packing: f32 -> f16, pad K to multiple of 32
// ---------------------------------------------------------------------------
__global__ void pack_x_f16(const float* __restrict__ X, _Float16* __restrict__ Xh,
                           int M, int Ksrc, int Kpad) {
  int i = blockIdx.x * blockDim.x + threadIdx.x;
  if (i >= M * Kpad) return;
  int r = i / Kpad, k = i - r * Kpad;
  Xh[i] = (_Float16)((k < Ksrc) ? X[(size_t)r * Ksrc + k] : 0.0f);
}

// Pack W [Ksrc x Nout] f32 row-major into WMMA B-fragment tiles:
// tile (nt, kt): 32 lanes x 16 halves contiguous.
// lane<16:  col = nt*16+lane,    K = kt*32 + (h<8 ? h : 16 + h-8)      (h=0..15)
// lane>=16: col = nt*16+lane-16, K = kt*32 + (h<8 ? 8 + h : 24 + h-8)
__global__ void pack_w_f16(const float* __restrict__ W, _Float16* __restrict__ Bp,
                           int Ksrc, int Kpad, int Nout) {
  int total = (Kpad >> 5) * (Nout >> 4) * 512;
  int i = blockIdx.x * blockDim.x + threadIdx.x;
  if (i >= total) return;
  int h    = i & 15;
  int lane = (i >> 4) & 31;
  int tile = i >> 9;
  int ktiles = Kpad >> 5;
  int kt = tile % ktiles;
  int nt = tile / ktiles;
  int col  = nt * 16 + (lane & 15);
  int krow = kt * 32 + ((h & 8) << 1) + ((lane >> 4) << 3) + (h & 7);
  float v = (krow < Ksrc) ? W[(size_t)krow * Nout + col] : 0.0f;
  Bp[i] = (_Float16)v;
}

// ---------------------------------------------------------------------------
// GEMM: C[M x Nout] = A[M x K](f16, K mult of 32) * Bp(packed f16) + bias
// One wave computes NT adjacent 16x16 output tiles (16 x 16*NT strip).
// Software-pipelined: fragment loads for k-step kt+1 are issued (one clause)
// before the NT WMMAs of k-step kt, so loads overlap matrix-pipe latency.
// ---------------------------------------------------------------------------
template <int NT>
__global__ void gemm_wmma(const _Float16* __restrict__ A,
                          const _Float16* __restrict__ Bp,
                          const float* __restrict__ bias,
                          float* __restrict__ C,
                          int M, int K, int Nout) {
  int wavesPerBlk = blockDim.x >> 5;
  int wave = blockIdx.x * wavesPerBlk + ((int)threadIdx.x >> 5);
  int lane = threadIdx.x & 31;
  int tnn = Nout / (16 * NT);              // N-tile groups per row strip
  int tilesTotal = (M >> 4) * tnn;
  if (wave >= tilesTotal) return;          // wave-uniform: EXEC stays all-1s
  int mt  = wave / tnn;
  int ntg = wave - mt * tnn;               // group of NT adjacent n-tiles

  // A fragment addressing (16x32 f16 layout)
  int kb = (lane >> 4) << 3;               // 0 or 8
  const _Float16* Aptr = A + (size_t)(mt * 16 + (lane & 15)) * K + kb;

  // B fragment addressing (pre-packed per-lane-contiguous tiles)
  int ktiles = K >> 5;
  const _Float16* Bbase[NT];
#pragma unroll
  for (int j = 0; j < NT; ++j)
    Bbase[j] = Bp + ((size_t)(ntg * NT + j) * ktiles * 32 + lane) * 16;

  v8f acc[NT];
#pragma unroll
  for (int j = 0; j < NT; ++j) acc[j] = (v8f){};

  // --- prologue: load fragments for kt = 0 ---
  v16h a_cur, b_cur[NT];
  {
    v8h alo = *(const v8h*)(Aptr);
    v8h ahi = *(const v8h*)(Aptr + 16);
    a_cur = __builtin_shufflevector(alo, ahi,
                                    0,1,2,3,4,5,6,7,8,9,10,11,12,13,14,15);
#pragma unroll
    for (int j = 0; j < NT; ++j) b_cur[j] = *(const v16h*)(Bbase[j]);
  }

  // --- main loop: load kt+1 while computing kt ---
  for (int kt = 0; kt < ktiles - 1; ++kt) {
    int kk = (kt + 1) << 5;
    v8h alo = *(const v8h*)(Aptr + kk);
    v8h ahi = *(const v8h*)(Aptr + kk + 16);
    v16h a_nxt = __builtin_shufflevector(alo, ahi,
                                         0,1,2,3,4,5,6,7,8,9,10,11,12,13,14,15);
    v16h b_nxt[NT];
#pragma unroll
    for (int j = 0; j < NT; ++j)
      b_nxt[j] = *(const v16h*)(Bbase[j] + (size_t)(kt + 1) * 512);
    if (kt + 2 < ktiles)
      __builtin_prefetch(Aptr + kk + 32, 0, 1);
#pragma unroll
    for (int j = 0; j < NT; ++j)
      acc[j] = __builtin_amdgcn_wmma_f32_16x16x32_f16(
          false, a_cur, false, b_cur[j], (short)0, acc[j], false, false);
    a_cur = a_nxt;
#pragma unroll
    for (int j = 0; j < NT; ++j) b_cur[j] = b_nxt[j];
  }

  // --- epilogue: last k-step ---
#pragma unroll
  for (int j = 0; j < NT; ++j)
    acc[j] = __builtin_amdgcn_wmma_f32_16x16x32_f16(
        false, a_cur, false, b_cur[j], (short)0, acc[j], false, false);

  int rbase = mt * 16 + ((lane >> 4) << 3);
#pragma unroll
  for (int j = 0; j < NT; ++j) {
    int col = (ntg * NT + j) * 16 + (lane & 15);
    float bv = bias[col];
#pragma unroll
    for (int v = 0; v < 8; ++v)
      C[(size_t)(rbase + v) * Nout + col] = acc[j][v] + bv;
  }
}

// ---------------------------------------------------------------------------
// Edge-phase kernels (self-loops appended logically as edges [E, E+N))
// ---------------------------------------------------------------------------
__device__ __forceinline__ void edge_sd(const int* __restrict__ ei, int E, int e,
                                        int& s, int& d) {
  if (e < E) { s = ei[e]; d = ei[E + e]; }
  else       { s = e - E; d = e - E; }
}

__device__ __forceinline__ unsigned f2ord(float f) {
  unsigned u = __float_as_uint(f);
  return (u & 0x80000000u) ? ~u : (u | 0x80000000u);
}
__device__ __forceinline__ float ord2f(unsigned u) {
  return __uint_as_float((u & 0x80000000u) ? (u & 0x7fffffffu) : ~u);
}

__global__ void edge_logits_max(const int* __restrict__ ei, int E, int ET, int H,
                                const float* __restrict__ xl,
                                const float* __restrict__ xr,
                                const float* __restrict__ att,
                                float* __restrict__ logit,
                                unsigned* __restrict__ mx) {
  int idx = blockIdx.x * blockDim.x + threadIdx.x;
  if (idx >= ET * H) return;
  int e = idx / H, h = idx - e * H;
  int s, d; edge_sd(ei, E, e, s, d);
  int HD = H * 32;
  const float4* pl = (const float4*)(xl + (size_t)s * HD + h * 32);
  const float4* pr = (const float4*)(xr + (size_t)d * HD + h * 32);
  const float4* pa = (const float4*)(att + h * 32);
  float acc = 0.f;
#pragma unroll
  for (int i = 0; i < 8; ++i) {
    float4 l = pl[i], r = pr[i], w = pa[i];
    float v;
    v = l.x + r.x; v = (v > 0.f) ? v : NEG_SLOPE * v; acc += v * w.x;
    v = l.y + r.y; v = (v > 0.f) ? v : NEG_SLOPE * v; acc += v * w.y;
    v = l.z + r.z; v = (v > 0.f) ? v : NEG_SLOPE * v; acc += v * w.z;
    v = l.w + r.w; v = (v > 0.f) ? v : NEG_SLOPE * v; acc += v * w.w;
  }
  logit[idx] = acc;
  atomicMax(mx + (size_t)d * H + h, f2ord(acc));
}

__global__ void edge_expsum(const int* __restrict__ ei, int E, int ET, int H,
                            float* __restrict__ logit,
                            const unsigned* __restrict__ mx,
                            float* __restrict__ denom) {
  int idx = blockIdx.x * blockDim.x + threadIdx.x;
  if (idx >= ET * H) return;
  int e = idx / H, h = idx - e * H;
  int s, d; edge_sd(ei, E, e, s, d);
  float a = __expf(logit[idx] - ord2f(mx[(size_t)d * H + h]));
  logit[idx] = a;
  atomicAdd(denom + (size_t)d * H + h, a);
}

__global__ void edge_aggregate(const int* __restrict__ ei, int E, int ET, int H,
                               const float* __restrict__ a,
                               const float* __restrict__ denom,
                               const float* __restrict__ xl,
                               float* __restrict__ out) {
  int idx = blockIdx.x * blockDim.x + threadIdx.x;
  if (idx >= ET * H) return;
  int e = idx / H, h = idx - e * H;
  int s, d; edge_sd(ei, E, e, s, d);
  float alpha = a[idx] / denom[(size_t)d * H + h];
  int HD = H * 32;
  const float4* ps = (const float4*)(xl + (size_t)s * HD + h * 32);
  float* po = out + (size_t)d * HD + h * 32;
#pragma unroll
  for (int i = 0; i < 8; ++i) {
    float4 v = ps[i];
    atomicAdd(po + 4 * i + 0, alpha * v.x);
    atomicAdd(po + 4 * i + 1, alpha * v.y);
    atomicAdd(po + 4 * i + 2, alpha * v.z);
    atomicAdd(po + 4 * i + 3, alpha * v.w);
  }
}

// ---------------------------------------------------------------------------
// Post: +bias, LayerNorm, ELU, optional residual; wave-per-node (wave32)
// Also re-emits f16 input for the next layer's WMMA GEMM.
// ---------------------------------------------------------------------------
__global__ void post_ln_elu(const float* __restrict__ accum,
                            const float* __restrict__ bo,
                            const float* __restrict__ g,
                            const float* __restrict__ be,
                            const float* __restrict__ resid,
                            float* __restrict__ hout,
                            _Float16* __restrict__ xh,
                            int Nn, int ld) {
  int wave = blockIdx.x * (blockDim.x >> 5) + ((int)threadIdx.x >> 5);
  int lane = threadIdx.x & 31;
  if (wave >= Nn) return;
  int cnt = ld >> 5;                       // 4 (ld=128) or 1 (ld=32)
  float v[4];
  float s = 0.f, s2 = 0.f;
  for (int i = 0; i < cnt; ++i) {
    int dm = lane + (i << 5);
    float t = accum[(size_t)wave * ld + dm] + bo[dm];
    v[i] = t; s += t; s2 += t * t;
  }
  for (int m = 16; m; m >>= 1) {
    s  += __shfl_xor(s,  m, 32);
    s2 += __shfl_xor(s2, m, 32);
  }
  float mu  = s / (float)ld;
  float var = s2 / (float)ld - mu * mu;
  float rs  = rsqrtf(var + LN_EPS);
  for (int i = 0; i < cnt; ++i) {
    int dm = lane + (i << 5);
    float y = (v[i] - mu) * rs * g[dm] + be[dm];
    y = (y > 0.f) ? y : (__expf(y) - 1.f);             // ELU
    if (resid) y += resid[(size_t)wave * ld + dm];     // post-ELU residual
    hout[(size_t)wave * ld + dm] = y;
    if (xh) xh[(size_t)wave * ld + dm] = (_Float16)y;
  }
}

// ---------------------------------------------------------------------------
// Head MLP: 32 -> 16 (ELU) -> 1
// ---------------------------------------------------------------------------
__global__ void head_mlp(const float* __restrict__ h,
                         const float* __restrict__ W1, const float* __restrict__ b1,
                         const float* __restrict__ W2, const float* __restrict__ b2,
                         float* __restrict__ out, int Nn) {
  int n = blockIdx.x * blockDim.x + threadIdx.x;
  if (n >= Nn) return;
  const float* hp = h + (size_t)n * 32;
  float acc2 = b2[0];
#pragma unroll
  for (int j = 0; j < 16; ++j) {
    float t = b1[j];
#pragma unroll
    for (int d2 = 0; d2 < 32; ++d2) t += hp[d2] * W1[d2 * 16 + j];
    t = (t > 0.f) ? t : (__expf(t) - 1.f);
    acc2 += t * W2[j];
  }
  out[n] = acc2;
}

// ---------------------------------------------------------------------------
extern "C" void kernel_launch(void* const* d_in, const int* in_sizes, int n_in,
                              void* d_out, int out_size, void* d_ws, size_t ws_size,
                              hipStream_t stream) {
  (void)n_in; (void)out_size; (void)ws_size;
  const float* x  = (const float*)d_in[0];
  const int*   ei = (const int*)d_in[1];
  int Nn = in_sizes[0] / 18;
  int E  = in_sizes[1] / 2;
  int ET = E + Nn;

  const float* Wl[3]  = {(const float*)d_in[2],  (const float*)d_in[10], (const float*)d_in[18]};
  const float* bl[3]  = {(const float*)d_in[3],  (const float*)d_in[11], (const float*)d_in[19]};
  const float* Wr[3]  = {(const float*)d_in[4],  (const float*)d_in[12], (const float*)d_in[20]};
  const float* br[3]  = {(const float*)d_in[5],  (const float*)d_in[13], (const float*)d_in[21]};
  const float* att[3] = {(const float*)d_in[6],  (const float*)d_in[14], (const float*)d_in[22]};
  const float* bo[3]  = {(const float*)d_in[7],  (const float*)d_in[15], (const float*)d_in[23]};
  const float* gg[3]  = {(const float*)d_in[8],  (const float*)d_in[16], (const float*)d_in[24]};
  const float* be[3]  = {(const float*)d_in[9],  (const float*)d_in[17], (const float*)d_in[25]};
  const float* cW1 = (const float*)d_in[26];
  const float* cb1 = (const float*)d_in[27];
  const float* cW2 = (const float*)d_in[28];
  const float* cb2 = (const float*)d_in[29];

  int Ksrc[3] = {18, 128, 128};
  int Kpad[3] = {32, 128, 128};
  int Nout[3] = {128, 128, 32};
  int Hh[3]   = {4, 4, 1};

  // ---- carve workspace ----
  char* ws = (char*)d_ws;
  size_t off = 0;
  auto carve = [&](size_t bytes) -> char* {
    char* p = ws + off;
    off += (bytes + 255) & ~(size_t)255;
    return p;
  };
  _Float16* Xh    = (_Float16*)carve((size_t)Nn * 128 * 2);
  float*    xl    = (float*)carve((size_t)Nn * 128 * 4);
  float*    xr    = (float*)carve((size_t)Nn * 128 * 4);
  float*    logit = (float*)carve((size_t)ET * 4 * 4);
  unsigned* mx    = (unsigned*)carve((size_t)Nn * 4 * 4);
  float*    denom = (float*)carve((size_t)Nn * 4 * 4);
  float*    accum = (float*)carve((size_t)Nn * 128 * 4);
  float*    h0    = (float*)carve((size_t)Nn * 128 * 4);
  float*    h1    = (float*)carve((size_t)Nn * 128 * 4);
  float*    h2    = (float*)carve((size_t)Nn * 32 * 4);
  _Float16* BpL[3];
  _Float16* BpR[3];
  for (int l = 0; l < 3; ++l) {
    size_t bytes = (size_t)(Kpad[l] >> 5) * (Nout[l] >> 4) * 512 * 2;
    BpL[l] = (_Float16*)carve(bytes);
    BpR[l] = (_Float16*)carve(bytes);
  }
  float* houts[3] = {h0, h1, h2};

  // ---- pack weights + input ----
  for (int l = 0; l < 3; ++l) {
    int tot = (Kpad[l] >> 5) * (Nout[l] >> 4) * 512;
    pack_w_f16<<<(tot + 255) / 256, 256, 0, stream>>>(Wl[l], BpL[l], Ksrc[l], Kpad[l], Nout[l]);
    pack_w_f16<<<(tot + 255) / 256, 256, 0, stream>>>(Wr[l], BpR[l], Ksrc[l], Kpad[l], Nout[l]);
  }
  {
    int tot = Nn * 32;
    pack_x_f16<<<(tot + 255) / 256, 256, 0, stream>>>(x, Xh, Nn, 18, 32);
  }

  // ---- GATv2 layers ----
  for (int l = 0; l < 3; ++l) {
    int K = Kpad[l], No = Nout[l], H = Hh[l];

    // Wave computes a 16 x 16*NT strip; Nn = 50000 = 3125*16, exact tiling.
    if (No == 128) {
      int tiles  = (Nn >> 4) * (No / 64);   // NT = 4
      int blocks = (tiles + 7) / 8;
      gemm_wmma<4><<<blocks, 256, 0, stream>>>(Xh, BpL[l], bl[l], xl, Nn, K, No);
      gemm_wmma<4><<<blocks, 256, 0, stream>>>(Xh, BpR[l], br[l], xr, Nn, K, No);
    } else {
      int tiles  = (Nn >> 4) * (No / 32);   // NT = 2
      int blocks = (tiles + 7) / 8;
      gemm_wmma<2><<<blocks, 256, 0, stream>>>(Xh, BpL[l], bl[l], xl, Nn, K, No);
      gemm_wmma<2><<<blocks, 256, 0, stream>>>(Xh, BpR[l], br[l], xr, Nn, K, No);
    }

    (void)hipMemsetAsync(mx,    0, (size_t)Nn * H * 4, stream);
    (void)hipMemsetAsync(denom, 0, (size_t)Nn * H * 4, stream);
    (void)hipMemsetAsync(accum, 0, (size_t)Nn * No * 4, stream);

    int tot = ET * H;
    edge_logits_max<<<(tot + 255) / 256, 256, 0, stream>>>(ei, E, ET, H, xl, xr, att[l], logit, mx);
    edge_expsum   <<<(tot + 255) / 256, 256, 0, stream>>>(ei, E, ET, H, logit, mx, denom);
    edge_aggregate<<<(tot + 255) / 256, 256, 0, stream>>>(ei, E, ET, H, logit, denom, xl, accum);

    const float* resid   = (l == 1) ? h0 : nullptr;
    _Float16*    xh_next = (l < 2) ? Xh : nullptr;
    post_ln_elu<<<(Nn + 7) / 8, 256, 0, stream>>>(accum, bo[l], gg[l], be[l],
                                                  resid, houts[l], xh_next, Nn, No);
  }

  // ---- head MLP ----
  head_mlp<<<(Nn + 255) / 256, 256, 0, stream>>>(h2, cW1, cb1, cW2, cb2,
                                                 (float*)d_out, Nn);
}
